// FrameSelector_6897717477505
// MI455X (gfx1250) — matrix operand order repaired and checked by reading, not compiled, not run on hardware
//
#include <hip/hip_runtime.h>
#include <hip/hip_bf16.h>

// Problem constants (match reference)
#define HID   1024
#define FEAT  2048
#define BATCH 64
#define SEQ   512
#define G4    (4 * HID)

#define BSTRIDE 40   // LDS row pitch in bf16 (80B): 16B-aligned, conflict-free across 16 lanes

typedef __attribute__((ext_vector_type(16))) __bf16       bf16x16;
typedef __attribute__((ext_vector_type(8)))  float        f32x8;
typedef __attribute__((ext_vector_type(4)))  unsigned int uintx4;
typedef __attribute__((ext_vector_type(2)))  unsigned int uintx2;

// ---------- async global->LDS path (guarded) ----------
#if __has_builtin(__builtin_amdgcn_global_load_async_to_lds_b128)
#define HAVE_ASYNC_LDS 1
#else
#define HAVE_ASYNC_LDS 0
#endif

#if HAVE_ASYNC_LDS
// Builtin signature (from clang diagnostic): arg0 is
//   __attribute__((__vector_size__(16))) int __device__ *   (addrspace(1), non-const)
typedef __attribute__((__vector_size__(4 * sizeof(int)))) int v4i;
typedef __attribute__((address_space(1))) v4i* gv4i_p;
typedef __attribute__((address_space(3))) v4i* lv4i_p;
#endif

static __device__ __forceinline__ void copy32_g2l(const __bf16* g, __bf16* l) {
#if HAVE_ASYNC_LDS
    __builtin_amdgcn_global_load_async_to_lds_b128((gv4i_p)(g),     (lv4i_p)(l),     0, 0);
    __builtin_amdgcn_global_load_async_to_lds_b128((gv4i_p)(g + 8), (lv4i_p)(l + 8), 0, 0);
#else
    uintx4 t0 = *(const uintx4*)g;
    uintx4 t1 = *(const uintx4*)(g + 8);
    *(uintx4*)l = t0;
    *(uintx4*)(l + 8) = t1;
#endif
}

static __device__ __forceinline__ void wait_async_keep2() {
#if HAVE_ASYNC_LDS
#if __has_builtin(__builtin_amdgcn_s_wait_asynccnt)
    __builtin_amdgcn_s_wait_asynccnt(2);
#else
    asm volatile("s_wait_asynccnt 0x2" ::: "memory");
#endif
#endif
}
static __device__ __forceinline__ void wait_async_all() {
#if HAVE_ASYNC_LDS
#if __has_builtin(__builtin_amdgcn_s_wait_asynccnt)
    __builtin_amdgcn_s_wait_asynccnt(0);
#else
    asm volatile("s_wait_asynccnt 0x0" ::: "memory");
#endif
#endif
}

// ---------- small helpers ----------
static __device__ __forceinline__ unsigned rnd_bf(unsigned u) {
    return (u + 0x7FFFu + ((u >> 16) & 1u)) >> 16;   // RNE fp32->bf16 on bit pattern
}
static __device__ __forceinline__ __bf16 f2bf(float f) {
    unsigned short h = (unsigned short)rnd_bf(__builtin_bit_cast(unsigned, f));
    return __builtin_bit_cast(__bf16, h);
}
static __device__ __forceinline__ float bf2f(__bf16 b) {
    unsigned u = ((unsigned)__builtin_bit_cast(unsigned short, b)) << 16;
    return __builtin_bit_cast(float, u);
}
static __device__ __forceinline__ float sigf(float x) {
    return 1.0f / (1.0f + __expf(-x));
}

union Frag { uintx4 q[2]; bf16x16 v; };

// A fragment (16x32 bf16) from row-major A, leading dim lda.
// lanes 0-15: M=lane, elems[0..7]=K(kg*8..), elems[8..15]=K(16+kg*8..)
static __device__ __forceinline__ bf16x16 load_a_bf16(const __bf16* __restrict__ A,
                                                      int lda, int m0, int k0, int lane) {
    int row = m0 + (lane & 15);
    int kg  = (lane >> 4) & 1;
    const __bf16* p = A + (size_t)row * lda + k0 + kg * 8;
    Frag f;
    f.q[0] = *(const uintx4*)(p);
    f.q[1] = *(const uintx4*)(p + 16);
    return f.v;
}

// B fragment (32x16 bf16) from LDS slab (rows at BSTRIDE pitch, 32 bf16 of K each)
static __device__ __forceinline__ bf16x16 load_b_lds(const __bf16* lds, int trow, int lane) {
    int r  = trow + (lane & 15);
    int kg = (lane >> 4) & 1;
    const __bf16* p = lds + r * BSTRIDE + kg * 8;
    Frag f;
    f.q[0] = *(const uintx4*)(p);
    f.q[1] = *(const uintx4*)(p + 16);
    return f.v;
}

static __device__ __forceinline__ f32x8 wmma_bf16(bf16x16 a, bf16x16 b, f32x8 c) {
    return __builtin_amdgcn_wmma_f32_16x16x32_bf16(false, a, false, b, (short)0, c, false, false);
}

// ---------- elementwise kernels ----------

// vectorized fp32 -> bf16 (n must be a multiple of 4; all callers satisfy this)
__global__ void f32_to_bf16_kernel(const float* __restrict__ src,
                                   __bf16* __restrict__ dst, int n) {
    int i = (blockIdx.x * blockDim.x + threadIdx.x) << 2;
    if (i >= n) return;
    uintx4 u = *(const uintx4*)(src + i);
    uintx2 o;
    o[0] = rnd_bf(u[0]) | (rnd_bf(u[1]) << 16);
    o[1] = rnd_bf(u[2]) | (rnd_bf(u[3]) << 16);
    *(uintx2*)(dst + i) = o;
}

__global__ void zero_bf16_kernel(__bf16* __restrict__ dst, int n) {
    int i = blockIdx.x * blockDim.x + threadIdx.x;
    if (i < n) dst[i] = __builtin_bit_cast(__bf16, (unsigned short)0);
}

// ---------- GEMM kernels ----------
// Both GEMMs: C(64 rows x 64 cols per block) = A @ W^T, K-loop double-buffered
// through LDS with async global->LDS staging of the shared B slab (64xK32).

// xe[r, h] = sum_f xbf[r, f] * Ew[h, f] + eb[h]   (r = b*SEQ + s)
// grid (512, 16), block 128 (4 waves; wave = 16-row M tile)
__global__ __launch_bounds__(128)
void embed_gemm_kernel(const __bf16* __restrict__ xbf, const __bf16* __restrict__ Ew,
                       const float* __restrict__ eb, __bf16* __restrict__ xe) {
    __shared__ __align__(16) __bf16 Bsm[2][64 * BSTRIDE];
    const int tid  = threadIdx.x;
    const int lane = tid & 31;
    const int wave = tid >> 5;
    const int m0 = blockIdx.x * 64 + wave * 16;
    const int n0 = blockIdx.y * 64;
    const int srow = tid >> 1;          // 0..63 staging row
    const int shalf = tid & 1;          // 16 bf16 halves
    const __bf16* gbase = Ew + (size_t)(n0 + srow) * FEAT + shalf * 16;

    f32x8 acc[4] = {};
    const int kiters = FEAT / 32;
    copy32_g2l(gbase, &Bsm[0][srow * BSTRIDE + shalf * 16]);
    for (int i = 0; i < kiters; ++i) {
        const int cur = i & 1;
        if (i + 1 < kiters) {
            copy32_g2l(gbase + (i + 1) * 32, &Bsm[cur ^ 1][srow * BSTRIDE + shalf * 16]);
            wait_async_keep2();
        } else {
            wait_async_all();
        }
        __syncthreads();
        bf16x16 a = load_a_bf16(xbf, FEAT, m0, i * 32, lane);
#pragma unroll
        for (int t = 0; t < 4; ++t) {
            bf16x16 b = load_b_lds(&Bsm[cur][0], t * 16, lane);
            acc[t] = wmma_bf16(a, b, acc[t]);
        }
        __syncthreads();
    }
    const int rbase = m0 + ((lane >> 4) << 3);
#pragma unroll
    for (int t = 0; t < 4; ++t) {
        int col = n0 + t * 16 + (lane & 15);
        float bias = eb[col];
#pragma unroll
        for (int v = 0; v < 8; ++v) {
            xe[(size_t)(rbase + v) * HID + col] = f2bf(acc[t][v] + bias);
        }
    }
}

// xg[(s*B + b), g] = sum_h xe[r, h] * Wih[g, h] + b_ih[g]  (stored (S,B,4H))
// grid (512, 64), block 128
__global__ __launch_bounds__(128)
void gates_gemm_kernel(const __bf16* __restrict__ xe, const __bf16* __restrict__ Wih,
                       const float* __restrict__ bih, float* __restrict__ xg) {
    __shared__ __align__(16) __bf16 Bsm[2][64 * BSTRIDE];
    const int tid  = threadIdx.x;
    const int lane = tid & 31;
    const int wave = tid >> 5;
    const int m0 = blockIdx.x * 64 + wave * 16;
    const int n0 = blockIdx.y * 64;
    const int srow = tid >> 1;
    const int shalf = tid & 1;
    const __bf16* gbase = Wih + (size_t)(n0 + srow) * HID + shalf * 16;

    f32x8 acc[4] = {};
    const int kiters = HID / 32;
    copy32_g2l(gbase, &Bsm[0][srow * BSTRIDE + shalf * 16]);
    for (int i = 0; i < kiters; ++i) {
        const int cur = i & 1;
        if (i + 1 < kiters) {
            copy32_g2l(gbase + (i + 1) * 32, &Bsm[cur ^ 1][srow * BSTRIDE + shalf * 16]);
            wait_async_keep2();
        } else {
            wait_async_all();
        }
        __syncthreads();
        bf16x16 a = load_a_bf16(xe, HID, m0, i * 32, lane);
#pragma unroll
        for (int t = 0; t < 4; ++t) {
            bf16x16 b = load_b_lds(&Bsm[cur][0], t * 16, lane);
            acc[t] = wmma_bf16(a, b, acc[t]);
        }
        __syncthreads();
    }
    const int rbase = m0 + ((lane >> 4) << 3);
#pragma unroll
    for (int t = 0; t < 4; ++t) {
        int col = n0 + t * 16 + (lane & 15);
        float bias = bih[col];
#pragma unroll
        for (int v = 0; v < 8; ++v) {
            int r = rbase + v;          // r = b*SEQ + s
            int bb = r >> 9;
            int ss = r & 511;
            xg[((size_t)ss * BATCH + bb) * G4 + col] = acc[t][v] + bias;
        }
    }
}

// ---------- LSTM recurrence ----------
// One timestep: gates = xg[s] + h_prev @ Whh^T + b_hh -> activations.
// grid 64 (hidden j-slices of 16), block 128 (4 waves = batch tiles of 16).
// The 4 gate slabs of Whh (4*16 rows x 32 K) are staged through LDS, shared by all waves.
__global__ __launch_bounds__(128)
void lstm_step_kernel(int s, const __bf16* __restrict__ hprev,
                      const __bf16* __restrict__ Whh, const float* __restrict__ bhh,
                      const float* __restrict__ xg, float* __restrict__ cbuf,
                      float* __restrict__ hout, __bf16* __restrict__ hnext,
                      __bf16* __restrict__ hhist) {
    __shared__ __align__(16) __bf16 Bsm[2][64 * BSTRIDE];
    const int tid  = threadIdx.x;
    const int lane = tid & 31;
    const int wave = tid >> 5;
    const int m0 = wave * 16;           // batch tile
    const int j0 = blockIdx.x * 16;     // hidden-unit slice
    const int srow = tid >> 1;          // 0..63 -> gate = srow>>4, col = srow&15
    const int shalf = tid & 1;
    const __bf16* gbase =
        Whh + (size_t)((srow >> 4) * HID + j0 + (srow & 15)) * HID + shalf * 16;

    f32x8 acc[4] = {};
    const int kiters = HID / 32;
    copy32_g2l(gbase, &Bsm[0][srow * BSTRIDE + shalf * 16]);
    for (int i = 0; i < kiters; ++i) {
        const int cur = i & 1;
        if (i + 1 < kiters) {
            copy32_g2l(gbase + (i + 1) * 32, &Bsm[cur ^ 1][srow * BSTRIDE + shalf * 16]);
            wait_async_keep2();
        } else {
            wait_async_all();
        }
        __syncthreads();
        bf16x16 a = load_a_bf16(hprev, HID, m0, i * 32, lane);
#pragma unroll
        for (int g = 0; g < 4; ++g) {
            bf16x16 b = load_b_lds(&Bsm[cur][0], g * 16, lane);
            acc[g] = wmma_bf16(a, b, acc[g]);
        }
        __syncthreads();
    }
    const int j = j0 + (lane & 15);
    const int rbase = m0 + ((lane >> 4) << 3);
    const float bi  = bhh[j];
    const float bf_ = bhh[HID + j];
    const float bg  = bhh[2 * HID + j];
    const float bo  = bhh[3 * HID + j];
#pragma unroll
    for (int v = 0; v < 8; ++v) {
        int rb = rbase + v;
        size_t xb = ((size_t)s * BATCH + rb) * G4;
        float iv = sigf(acc[0][v] + xg[xb + j] + bi);
        float fv = sigf(acc[1][v] + xg[xb + HID + j] + bf_);
        float gv = tanhf(acc[2][v] + xg[xb + 2 * HID + j] + bg);
        float ov = sigf(acc[3][v] + xg[xb + 3 * HID + j] + bo);
        size_t hidx = (size_t)rb * HID + j;
        float cold = (s == 0) ? 0.0f : cbuf[hidx];
        float cn = fv * cold + iv * gv;
        float hn = ov * tanhf(cn);
        cbuf[hidx] = cn;
        hout[hidx] = hn;
        __bf16 hb = f2bf(hn);
        hnext[hidx] = hb;
        hhist[((size_t)s * BATCH + rb) * HID + j] = hb;
    }
}

// logits[b, s] = dot(h_t[b], head_w) + head_b ; one wave per (s,b)
__global__ __launch_bounds__(256)
void head_kernel(const __bf16* __restrict__ hhist, const float* __restrict__ hw,
                 const float* __restrict__ hb, float* __restrict__ logits) {
    const int lane = threadIdx.x & 31;
    const int wave = threadIdx.x >> 5;
    const int r = blockIdx.x * 8 + wave;          // r = s*BATCH + b
    const __bf16* hp = hhist + (size_t)r * HID;
    float sum = 0.0f;
    for (int t = 0; t < HID; t += 32) sum += bf2f(hp[t + lane]) * hw[t + lane];
#pragma unroll
    for (int off = 16; off > 0; off >>= 1) sum += __shfl_down(sum, off, 32);
    if (lane == 0) {
        int ss = r >> 6;
        int bb = r & 63;
        logits[bb * SEQ + ss] = sum + hb[0];
    }
}

// ---------- host orchestration ----------
extern "C" void kernel_launch(void* const* d_in, const int* in_sizes, int n_in,
                              void* d_out, int out_size, void* d_ws, size_t ws_size,
                              hipStream_t stream) {
    (void)in_sizes; (void)n_in; (void)out_size; (void)ws_size;

    const float* x       = (const float*)d_in[0];
    const float* embed_w = (const float*)d_in[1];
    const float* embed_b = (const float*)d_in[2];
    const float* w_ih    = (const float*)d_in[3];
    const float* b_ih    = (const float*)d_in[4];
    const float* w_hh    = (const float*)d_in[5];
    const float* b_hh    = (const float*)d_in[6];
    const float* head_w  = (const float*)d_in[7];
    const float* head_b  = (const float*)d_in[8];

    char* ws = (char*)d_ws;
    size_t o = 0;
    __bf16* Ew    = (__bf16*)(ws + o); o += (size_t)HID * FEAT * 2;
    __bf16* Wih   = (__bf16*)(ws + o); o += (size_t)G4 * HID * 2;
    __bf16* Whh   = (__bf16*)(ws + o); o += (size_t)G4 * HID * 2;
    __bf16* xe    = (__bf16*)(ws + o); o += (size_t)BATCH * SEQ * HID * 2;
    float*  xg    = (float*)(ws + o);  o += (size_t)SEQ * BATCH * G4 * 4;
    __bf16* hbuf0 = (__bf16*)(ws + o); o += (size_t)BATCH * HID * 2;
    __bf16* hbuf1 = (__bf16*)(ws + o); o += (size_t)BATCH * HID * 2;
    __bf16* hhist = (__bf16*)(ws + o); o += (size_t)SEQ * BATCH * HID * 2;
    __bf16* xbf   = (__bf16*)(ws + o);

    float* out    = (float*)d_out;
    float* logits = out;                                 // (B, S)
    float* hout   = out + (size_t)BATCH * SEQ;           // (B, HID)
    float* cbuf   = out + (size_t)BATCH * SEQ + (size_t)BATCH * HID;  // (B, HID)

    // fp32 -> bf16 conversions (vectorized, memory-bound passes)
    {
        int n = BATCH * SEQ * FEAT;
        f32_to_bf16_kernel<<<(n / 4 + 255) / 256, 256, 0, stream>>>(x, xbf, n);
    }
    f32_to_bf16_kernel<<<(HID * FEAT / 4 + 255) / 256, 256, 0, stream>>>(embed_w, Ew, HID * FEAT);
    f32_to_bf16_kernel<<<(G4 * HID / 4 + 255) / 256, 256, 0, stream>>>(w_ih, Wih, G4 * HID);
    f32_to_bf16_kernel<<<(G4 * HID / 4 + 255) / 256, 256, 0, stream>>>(w_hh, Whh, G4 * HID);

    // big GEMMs (bf16 WMMA, LDS-staged B slabs)
    embed_gemm_kernel<<<dim3(512, 16), 128, 0, stream>>>(xbf, Ew, embed_b, xe);
    gates_gemm_kernel<<<dim3(512, 64), 128, 0, stream>>>(xe, Wih, b_ih, xg);

    // recurrence: 512 dependent steps
    zero_bf16_kernel<<<(BATCH * HID + 255) / 256, 256, 0, stream>>>(hbuf0, BATCH * HID);
    for (int s = 0; s < SEQ; ++s) {
        const __bf16* hp = (s & 1) ? hbuf1 : hbuf0;
        __bf16*       hn = (s & 1) ? hbuf0 : hbuf1;
        lstm_step_kernel<<<64, 128, 0, stream>>>(s, hp, Whh, b_hh, xg, cbuf, hout, hn, hhist);
    }

    // head
    head_kernel<<<(SEQ * BATCH) / 8, 256, 0, stream>>>(hhist, head_w, head_b, logits);
}
